// PNPPDS_69956427317651
// MI455X (gfx1250) — matrix operand order repaired and checked by reading, not compiled, not run on hardware
//
#include <hip/hip_runtime.h>
#include <hip/hip_bf16.h>

typedef __attribute__((ext_vector_type(16))) _Float16 v16h;
typedef __attribute__((ext_vector_type(8)))  _Float16 h8;
typedef __attribute__((ext_vector_type(8)))  float    v8f;

#define DEV __device__ __forceinline__

DEV v16h hzero(){ v16h v; 
#pragma unroll
  for (int i=0;i<16;i++) v[i]=(_Float16)0.f; return v; }
DEV v8f fzero(){ v8f v;
#pragma unroll
  for (int i=0;i<8;i++) v[i]=0.f; return v; }
DEV float actf(float x, float a){ return x>0.f ? x : a*x; }

// ---------------------------------------------------------------- utilities

__global__ void k_extract(const float* __restrict__ x, float* __restrict__ xs){
  // x: [B][4][3][N] -> xs: [t*B+b][N][3]   (B=2, N=8192)
  const int Nn=8192, Bn=2;
  int tid = blockIdx.x*blockDim.x + threadIdx.x;
  if (tid >= 8*Nn) return;
  int tb = tid / Nn, i = tid % Nn;
  int t = tb / Bn, b = tb % Bn;
#pragma unroll
  for (int c=0;c<3;c++)
    xs[((size_t)tb*Nn + i)*3 + c] = x[(((size_t)b*4 + t)*3 + c)*Nn + i];
}

__global__ void k_convert(const float* __restrict__ W, _Float16* __restrict__ Wt,
                          int cin, int cout, int kp){
  int tid = blockIdx.x*blockDim.x + threadIdx.x;
  if (tid >= cout*kp) return;
  int nn = tid / kp, kk = tid % kp;
  Wt[tid] = (kk < cin) ? (_Float16)W[(size_t)kk*cout + nn] : (_Float16)0.f;
}

__global__ void k_fps(const float* __restrict__ xyz_all, int n, int npoint,
                      int* __restrict__ idx_out){
  __shared__ float dist[8192];
  __shared__ float rv[256];
  __shared__ int   ri[256];
  const float* xyz = xyz_all + (size_t)blockIdx.x*n*3;
  int* idx = idx_out + (size_t)blockIdx.x*npoint;
  int t = threadIdx.x;
  for (int i=t;i<n;i+=256) dist[i]=1e10f;
  __syncthreads();
  int far = 0;
  for (int p=0;p<npoint;p++){
    if (t==0) idx[p]=far;
    float cx=xyz[(size_t)far*3+0], cy=xyz[(size_t)far*3+1], cz=xyz[(size_t)far*3+2];
    float bv=-1.f; int bi=0;
    for (int i=t;i<n;i+=256){
      float dx=xyz[(size_t)i*3+0]-cx, dy=xyz[(size_t)i*3+1]-cy, dz=xyz[(size_t)i*3+2]-cz;
      float d=dx*dx+dy*dy+dz*dz;
      float dm=fminf(dist[i],d);
      dist[i]=dm;
      if (dm>bv){bv=dm;bi=i;}
    }
    rv[t]=bv; ri[t]=bi;
    __syncthreads();
    for (int s=128;s>0;s>>=1){
      if (t<s){
        if (rv[t+s]>rv[t] || (rv[t+s]==rv[t] && ri[t+s]<ri[t])){ rv[t]=rv[t+s]; ri[t]=ri[t+s]; }
      }
      __syncthreads();
    }
    far = ri[0];
    __syncthreads();
  }
}

__global__ void k_gather_pos(const float* __restrict__ xyz, const int* __restrict__ idx,
                             float* __restrict__ out, int n, int npt, int total){
  int q = blockIdx.x*blockDim.x + threadIdx.x;
  if (q>=total) return;
  int cl = q/npt;
  int id = idx[q];
#pragma unroll
  for (int c=0;c<3;c++)
    out[(size_t)q*3+c] = xyz[((size_t)cl*n + id)*3 + c];
}

__global__ void k_ballq(const float* __restrict__ xyz, const float* __restrict__ nxyz,
                        int* __restrict__ gidx, int n, int S, float r2, int ns, int total){
  int q = blockIdx.x*blockDim.x + threadIdx.x;
  if (q>=total) return;
  int cl = q/S;
  const float* cand = xyz + (size_t)cl*n*3;
  float px=nxyz[(size_t)q*3], py=nxyz[(size_t)q*3+1], pz=nxyz[(size_t)q*3+2];
  int cnt=0, first=0;
  for (int c=0;c<n && cnt<ns;c++){
    float dx=cand[c*3]-px, dy=cand[c*3+1]-py, dz=cand[c*3+2]-pz;
    float d=dx*dx+dy*dy+dz*dz;
    if (d<=r2){ if(cnt==0) first=c; gidx[(size_t)q*ns+cnt]=c; cnt++; }
  }
  for (;cnt<ns;cnt++) gidx[(size_t)q*ns+cnt]=first;
}

__global__ void k_knn(const float* __restrict__ p2, const float* __restrict__ p1,
                      int* __restrict__ out, int n2, int n1, int k, int total){
  int q = blockIdx.x*blockDim.x + threadIdx.x;
  if (q>=total) return;
  int cl = q/n1;
  const float* cand = p2 + (size_t)cl*n2*3;
  float px=p1[(size_t)q*3], py=p1[(size_t)q*3+1], pz=p1[(size_t)q*3+2];
  float bd[24]; int bi[24];
  for (int i=0;i<24;i++){bd[i]=3.4e38f; bi[i]=0;}
  float wv=3.4e38f; int wx=0;
  for (int c=0;c<n2;c++){
    float dx=cand[c*3]-px, dy=cand[c*3+1]-py, dz=cand[c*3+2]-pz;
    float d=dx*dx+dy*dy+dz*dz;
    if (d < wv){
      bd[wx]=d; bi[wx]=c;
      float nv=-1.f; int nx=0;
      for (int i=0;i<k;i++){ if (bd[i]>nv){nv=bd[i]; nx=i;} }
      wv=nv; wx=nx;
    }
  }
  for (int i=0;i<k;i++) out[(size_t)q*k+i]=bi[i];
}

__global__ void k_concat2(const float* __restrict__ a, int wa,
                          const float* __restrict__ b, int wb,
                          float* __restrict__ dst, int total, int wtot){
  int tid = blockIdx.x*blockDim.x + threadIdx.x;
  if (tid>=total) return;
  int r = tid/wtot, c = tid%wtot;
  dst[tid] = (c<wa) ? a[(size_t)r*wa+c] : b[(size_t)r*wb + (c-wa)];
}

__global__ void k_fp_interp(const float* __restrict__ pos1, const float* __restrict__ pos2,
                            const float* __restrict__ f2, float* __restrict__ A,
                            int n1, int n2, int C, int total){
  int q = blockIdx.x*blockDim.x + threadIdx.x;
  if (q>=total) return;
  int cl = q/n1;
  const float* cand = pos2 + (size_t)cl*n2*3;
  float px=pos1[(size_t)q*3], py=pos1[(size_t)q*3+1], pz=pos1[(size_t)q*3+2];
  float d0=3.4e38f,d1=3.4e38f,d2=3.4e38f; int i0=0,i1=0,i2=0;
  for (int c=0;c<n2;c++){
    float dx=cand[c*3]-px, dy=cand[c*3+1]-py, dz=cand[c*3+2]-pz;
    float d=dx*dx+dy*dy+dz*dz;
    if (d<d0){ d2=d1;i2=i1; d1=d0;i1=i0; d0=d;i0=c; }
    else if (d<d1){ d2=d1;i2=i1; d1=d;i1=c; }
    else if (d<d2){ d2=d;i2=c; }
  }
  float w0=1.f/(d0+1e-8f), w1=1.f/(d1+1e-8f), w2=1.f/(d2+1e-8f);
  float s=w0+w1+w2; w0/=s; w1/=s; w2/=s;
  const float* F0=f2+(size_t)(cl*n2+i0)*C;
  const float* F1=f2+(size_t)(cl*n2+i1)*C;
  const float* F2=f2+(size_t)(cl*n2+i2)*C;
  float* dst = A + (size_t)q*(C+1);
  dst[0]=0.f;
  for (int c=0;c<C;c++) dst[1+c] = w0*F0[c]+w1*F1[c]+w2*F2[c];
}

__global__ void k_final(const float* __restrict__ x, const float* __restrict__ flow,
                        float* __restrict__ out, int total){
  const int Nn=8192;
  int tid = blockIdx.x*blockDim.x + threadIdx.x;
  if (tid>=total) return;
  int b = tid/(3*Nn); int rem = tid - b*3*Nn; int c = rem/Nn; int i = rem - c*Nn;
  out[tid] = x[((size_t)(b*4+3)*3 + c)*Nn + i] + flow[((size_t)b*Nn + i)*3 + c];
}

// ---------------------------------------------------------------- WMMA GEMMs
// One wave (32 threads) per block computes a 16-row x 64-col output strip.
// A fragment (16x32 f16): lane<16 -> K {0..7,16..23}, lane>=16 -> K {8..15,24..31}, row = lane&15
// B fragment (32x16 f16): col = lane&15, K same as A fragment; weights stored
//   transposed+padded as Wt[Cout][Kpad] so each lane's halves load as 2 x b128.

DEV v16h load_bfrag(const _Float16* __restrict__ Wt, int Kpad, int col, int Cout,
                    int k0, int kh){
  v16h bf = hzero();
  if (col < Cout){
    const h8* wp = (const h8*)(Wt + (size_t)col*Kpad + k0 + kh*8);
    h8 lo = wp[0], hi = wp[2];
#pragma unroll
    for (int i=0;i<8;i++){ bf[i]=lo[i]; bf[8+i]=hi[i]; }
  }
  return bf;
}

// Dense A tile load: uniform full-tile fast path (straight contiguous loads),
// guarded path only for the K tail tile.
DEV void load_a_dense(v16h& a, const float* __restrict__ ap, int K, int k0, int kh){
  if (k0 + 32 <= K){
    const float* p = ap + k0 + kh*8;
#pragma unroll
    for (int i=0;i<8;i++) a[i] = (_Float16)p[i];
#pragma unroll
    for (int i=0;i<8;i++) a[8+i] = (_Float16)p[16+i];
  } else {
    int kA = k0 + kh*8;
#pragma unroll
    for (int i=0;i<8;i++){ int k=kA+i; if (k<K) a[i] = (_Float16)ap[k]; }
    int kA2 = k0 + 16 + kh*8;
#pragma unroll
    for (int i=0;i<8;i++){ int k=kA2+i; if (k<K) a[8+i] = (_Float16)ap[k]; }
  }
}

// Dense GEMM: Out = act(A[MxK] @ W + bias); optional fused max-pool over
// groups of 16 consecutive rows (tile == group) when fuse_max16 != 0.
__global__ void k_gemm(const float* __restrict__ A, int M, int K,
                       const _Float16* __restrict__ Wt, int Kpad,
                       const float* __restrict__ bias, int Cout,
                       float* __restrict__ Out, float alpha, int fuse_max16){
  int lane = threadIdx.x;
  int rlo = lane & 15, kh = lane >> 4;
  int mbase = blockIdx.x * 16;
  int nb = blockIdx.y * 64;
  int arow = mbase + rlo;
  v8f acc[4] = {fzero(),fzero(),fzero(),fzero()};
  const float* ap = A + (size_t)(arow < M ? arow : 0)*K;
  bool rowok = arow < M;
  for (int k0=0;k0<Kpad;k0+=32){
    v16h a = hzero();
    if (rowok) load_a_dense(a, ap, K, k0, kh);
#pragma unroll
    for (int t=0;t<4;t++){
      v16h bf = load_bfrag(Wt, Kpad, nb + t*16 + rlo, Cout, k0, kh);
      acc[t] = __builtin_amdgcn_wmma_f32_16x16x32_f16(false, a, false, bf,
                                                      (short)0, acc[t], false, false);
    }
  }
  if (fuse_max16){
#pragma unroll
    for (int t=0;t<4;t++){
      int col = nb + t*16 + rlo;
      float bv = (col<Cout) ? bias[col] : 0.f;
      float mx = -3.4e38f;
#pragma unroll
      for (int v=0;v<8;v++) mx = fmaxf(mx, actf(acc[t][v]+bv, alpha));
      mx = fmaxf(mx, __shfl_xor(mx, 16, 32));
      if (kh==0 && col<Cout) Out[(size_t)blockIdx.x*Cout + col] = mx;
    }
  } else {
#pragma unroll
    for (int t=0;t<4;t++){
      int col = nb + t*16 + rlo;
      if (col<Cout){
        float bv = bias[col];
#pragma unroll
        for (int v=0;v<8;v++){
          int m = mbase + kh*8 + v;
          if (m < M) Out[(size_t)m*Cout + col] = actf(acc[t][v]+bv, alpha);
        }
      }
    }
  }
}

// On-the-fly grouped A-matrix: row = (group g, neighbor j); columns are
// [pos_diff | gathered feature | broadcast feature] segments.
struct GSpec {
  const float* pos1;   // [NG][3]          per-group center
  const float* pos2;   // [clouds][n2][3]  neighbor positions
  const int*   gidx;   // [NG][ns]         neighbor indices (into n2)
  const float* Fg;     // [clouds][n2][Cg] gathered features (nullptr -> zeros)
  const float* Fb;     // [NG][Cb]         broadcast features
  int S, n2, ns, Cg, Cb, pd_off, g_off, b_off;
};

DEV float gfetch(const GSpec& gs, int g, int nbrabs, int k, const float* p1row){
  unsigned kp = (unsigned)(k - gs.pd_off);
  if (kp < 3u) return gs.pos2[(size_t)nbrabs*3 + (int)kp] - p1row[(int)kp];
  unsigned kg = (unsigned)(k - gs.g_off);
  if (kg < (unsigned)gs.Cg) return gs.Fg ? gs.Fg[(size_t)nbrabs*gs.Cg + (int)kg] : 0.f;
  unsigned kb = (unsigned)(k - gs.b_off);
  if (kb < (unsigned)gs.Cb) return gs.Fb[(size_t)g*gs.Cb + (int)kb];
  return 0.f;
}

// Grouped A tile load with uniform fast paths: whole 32-wide K tile inside the
// gathered-feature segment or inside the broadcast segment -> straight loads.
DEV void load_a_grouped(v16h& a, const GSpec& gs, int g, int nbrabs, int k0, int kh,
                        const float* p1row){
  if (gs.Fg && k0 >= gs.g_off && (k0 + 32) <= (gs.g_off + gs.Cg)){
    const float* p = gs.Fg + (size_t)nbrabs*gs.Cg + (k0 - gs.g_off) + kh*8;
#pragma unroll
    for (int i=0;i<8;i++) a[i] = (_Float16)p[i];
#pragma unroll
    for (int i=0;i<8;i++) a[8+i] = (_Float16)p[16+i];
    return;
  }
  if (gs.Fb && k0 >= gs.b_off && (k0 + 32) <= (gs.b_off + gs.Cb)){
    const float* p = gs.Fb + (size_t)g*gs.Cb + (k0 - gs.b_off) + kh*8;
#pragma unroll
    for (int i=0;i<8;i++) a[i] = (_Float16)p[i];
#pragma unroll
    for (int i=0;i<8;i++) a[8+i] = (_Float16)p[16+i];
    return;
  }
  int kA = k0 + kh*8;
#pragma unroll
  for (int i=0;i<8;i++) a[i] = (_Float16)gfetch(gs, g, nbrabs, kA+i, p1row);
  int kA2 = k0 + 16 + kh*8;
#pragma unroll
  for (int i=0;i<8;i++) a[8+i] = (_Float16)gfetch(gs, g, nbrabs, kA2+i, p1row);
}

// Grouped GEMM, materializing all NG*ns rows (relu). Used before 2nd MLP layers.
__global__ void k_ggemm(GSpec gs, int NG, const _Float16* __restrict__ Wt, int Kpad,
                        const float* __restrict__ bias, int Cout, float* __restrict__ Out){
  int lane = threadIdx.x;
  int rlo = lane & 15, kh = lane >> 4;
  int mbase = blockIdx.x * 16;
  int nb = blockIdx.y * 64;
  int r = mbase + rlo;
  int Mtot = NG * gs.ns;
  bool rv = r < Mtot;
  int g=0, nbr=0; const float* p1row = gs.pos1;
  if (rv){
    g = r / gs.ns; int j = r - g*gs.ns;
    nbr = (g/gs.S)*gs.n2 + gs.gidx[(size_t)g*gs.ns + j];
    p1row = gs.pos1 + (size_t)g*3;
  }
  v8f acc[4] = {fzero(),fzero(),fzero(),fzero()};
  for (int k0=0;k0<Kpad;k0+=32){
    v16h a = hzero();
    if (rv) load_a_grouped(a, gs, g, nbr, k0, kh, p1row);
#pragma unroll
    for (int t=0;t<4;t++){
      v16h bf = load_bfrag(Wt, Kpad, nb + t*16 + rlo, Cout, k0, kh);
      acc[t] = __builtin_amdgcn_wmma_f32_16x16x32_f16(false, a, false, bf,
                                                      (short)0, acc[t], false, false);
    }
  }
#pragma unroll
  for (int t=0;t<4;t++){
    int col = nb + t*16 + rlo;
    if (col<Cout){
      float bv = bias[col];
#pragma unroll
      for (int v=0;v<8;v++){
        int m = mbase + kh*8 + v;
        if (m < Mtot) Out[(size_t)m*Cout + col] = fmaxf(acc[t][v]+bv, 0.f);
      }
    }
  }
}

// Grouped single-layer GEMM with fused max over neighbors (relu).
// One block per group; ns handled as ceil(ns/16) masked 16-row sub-tiles.
__global__ void k_ggemm_max(GSpec gs, int NG, const _Float16* __restrict__ Wt, int Kpad,
                            const float* __restrict__ bias, int Cout, float* __restrict__ Out){
  int lane = threadIdx.x;
  int rlo = lane & 15, kh = lane >> 4;
  int g = blockIdx.x;
  int nb = blockIdx.y * 64;
  int cl = g / gs.S;
  const float* p1row = gs.pos1 + (size_t)g*3;
  float mx[4] = {-3.4e38f,-3.4e38f,-3.4e38f,-3.4e38f};
  int nsub = (gs.ns + 15) >> 4;
  for (int jt=0; jt<nsub; jt++){
    int j = jt*16 + rlo;
    bool jv = j < gs.ns;
    int nbr = jv ? (cl*gs.n2 + gs.gidx[(size_t)g*gs.ns + j]) : 0;
    v8f acc[4] = {fzero(),fzero(),fzero(),fzero()};
    for (int k0=0;k0<Kpad;k0+=32){
      v16h a = hzero();
      if (jv) load_a_grouped(a, gs, g, nbr, k0, kh, p1row);
#pragma unroll
      for (int t=0;t<4;t++){
        v16h bf = load_bfrag(Wt, Kpad, nb + t*16 + rlo, Cout, k0, kh);
        acc[t] = __builtin_amdgcn_wmma_f32_16x16x32_f16(false, a, false, bf,
                                                        (short)0, acc[t], false, false);
      }
    }
#pragma unroll
    for (int t=0;t<4;t++){
      int col = nb + t*16 + rlo;
      float bv = (col<Cout) ? bias[col] : 0.f;
#pragma unroll
      for (int v=0;v<8;v++){
        int jc = jt*16 + kh*8 + v;
        if (jc < gs.ns) mx[t] = fmaxf(mx[t], fmaxf(acc[t][v]+bv, 0.f));
      }
    }
  }
#pragma unroll
  for (int t=0;t<4;t++){
    int col = nb + t*16 + rlo;
    float m2 = fmaxf(mx[t], __shfl_xor(mx[t], 16, 32));
    if (kh==0 && col<Cout) Out[(size_t)g*Cout + col] = m2;
  }
}

// ---------------------------------------------------------------- host

extern "C" void kernel_launch(void* const* d_in, const int* in_sizes, int n_in,
                              void* d_out, int out_size, void* d_ws, size_t ws_size,
                              hipStream_t stream) {
  (void)in_sizes; (void)n_in; (void)out_size; (void)ws_size;
  const float* x = (const float*)d_in[0];
  auto F = [&](int i){ return (const float*)d_in[i]; };

  char* base = (char*)d_ws;
  size_t off = 0;
  auto alloc = [&](size_t bytes)->char*{
    char* p = base + off;
    off += (bytes + 255) & ~(size_t)255;
    return p;
  };

  constexpr int Bn=2, Nn=8192, S1=5500, S2=1375, S3=275;
  auto cdiv = [](int a, int b){ return (a+b-1)/b; };

  float* xs    = (float*)alloc((size_t)8*Nn*3*4);
  float* GA    = (float*)alloc((size_t)22528000*4);   // big staging buffer
  float* l1pos = (float*)alloc((size_t)4*Bn*S1*3*4);
  float* l1ft  = (float*)alloc((size_t)4*Bn*S1*128*4);
  float* fb    = (float*)alloc((size_t)2*Bn*S1*128*4);
  float* p2pos = (float*)alloc((size_t)2*Bn*S2*3*4);
  float* f2a   = (float*)alloc((size_t)2*Bn*S2*256*4);
  float* f2b   = (float*)alloc((size_t)Bn*S2*256*4);
  float* p3pos = (float*)alloc((size_t)Bn*S3*3*4);
  float* f3    = (float*)alloc((size_t)Bn*S3*512*4);
  float* hbuf  = (float*)alloc((size_t)Bn*S1*512*4);
  float* f2n   = (float*)alloc((size_t)Bn*S2*512*4);
  float* f1n   = (float*)alloc((size_t)Bn*S1*512*4);
  float* f0b   = (float*)alloc((size_t)Bn*Nn*256*4);
  float* h1b   = (float*)alloc((size_t)Bn*Nn*128*4);
  float* flow  = (float*)alloc((size_t)Bn*Nn*3*4);
  int* fps1  = (int*)alloc((size_t)8*S1*4);
  int* gidx1 = (int*)alloc((size_t)8*S1*16*4);
  int* knn1  = (int*)alloc((size_t)2*Bn*S1*24*4);
  int* fps2  = (int*)alloc((size_t)2*Bn*S2*4);
  int* gidx2 = (int*)alloc((size_t)2*Bn*S2*16*4);
  int* knn2  = (int*)alloc((size_t)Bn*S2*24*4);
  int* fps3  = (int*)alloc((size_t)Bn*S3*4);
  int* gidx3 = (int*)alloc((size_t)Bn*S3*16*4);
  int* knnu1 = (int*)alloc((size_t)Bn*S2*16*4);
  int* knnu2 = (int*)alloc((size_t)Bn*S1*16*4);

  // weight conversion: f32 [Cin][Cout] -> f16 transposed+padded [Cout][Kpad]
  struct LDesc { int wi, bi, cin, cout; };
  const LDesc ld[14] = {
    {1,2,4,128},{3,4,128,128},{5,6,259,128},{7,8,131,256},{9,10,256,256},
    {11,12,515,256},{13,14,259,512},{15,16,515,512},{17,18,768,512},
    {19,20,515,512},{21,22,640,512},{23,24,513,256},{25,26,256,128},{27,28,128,3}
  };
  _Float16* wt[14]; int kpad[14];
  for (int l=0;l<14;l++){
    kpad[l] = ((ld[l].cin + 31)/32)*32;
    wt[l] = (_Float16*)alloc((size_t)ld[l].cout*kpad[l]*2);
    int tot = ld[l].cout*kpad[l];
    k_convert<<<cdiv(tot,256),256,0,stream>>>(F(ld[l].wi), wt[l], ld[l].cin, ld[l].cout, kpad[l]);
  }

  // 1) split frames into [cloud][N][3]
  k_extract<<<cdiv(8*Nn,256),256,0,stream>>>(x, xs);

  // 2) level-1 FPS + ball query (8 clouds at once)
  k_fps<<<8,256,0,stream>>>(xs, Nn, S1, fps1);
  k_gather_pos<<<cdiv(8*S1,256),256,0,stream>>>(xs, fps1, l1pos, Nn, S1, 8*S1);
  k_ballq<<<cdiv(8*S1,256),256,0,stream>>>(xs, l1pos, gidx1, Nn, S1, 0.25f, 16, 8*S1);

  // 3) sa1 per frame: [4,128] grouped gemm -> [128,128] gemm + fused max16
  for (int t=0;t<4;t++){
    GSpec gs{};
    gs.pos1 = l1pos + (size_t)t*Bn*S1*3;
    gs.pos2 = xs + (size_t)t*Bn*Nn*3;
    gs.gidx = gidx1 + (size_t)t*Bn*S1*16;
    gs.Fg = nullptr; gs.Fb = nullptr;
    gs.S=S1; gs.n2=Nn; gs.ns=16; gs.Cg=1; gs.Cb=0;
    gs.pd_off=0; gs.g_off=3; gs.b_off=0x40000000;
    k_ggemm<<<dim3(Bn*S1,2),32,0,stream>>>(gs, Bn*S1, wt[0], kpad[0], F(2), 128, GA);
    k_gemm<<<dim3(Bn*S1,2),32,0,stream>>>(GA, Bn*S1*16, 128, wt[1], kpad[1], F(4), 128,
                                          l1ft + (size_t)t*Bn*S1*128, 0.f, 1);
  }

  // 4) fe1 on frame pairs (1,0) and (3,2): knn24 + fused grouped gemm-max
  for (int p=0;p<2;p++){
    int t1 = 2*p+1, t0 = 2*p;
    const float* P1 = l1pos + (size_t)t1*Bn*S1*3;
    const float* P0 = l1pos + (size_t)t0*Bn*S1*3;
    int* ki = knn1 + (size_t)p*Bn*S1*24;
    k_knn<<<cdiv(Bn*S1,256),256,0,stream>>>(P0, P1, ki, S1, S1, 24, Bn*S1);
    GSpec gs{};
    gs.pos1=P1; gs.pos2=P0; gs.gidx=ki;
    gs.Fg = l1ft + (size_t)t0*Bn*S1*128; gs.Fb = l1ft + (size_t)t1*Bn*S1*128;
    gs.S=S1; gs.n2=S1; gs.ns=24; gs.Cg=128; gs.Cb=128;
    gs.pd_off=0; gs.g_off=3; gs.b_off=131;
    k_ggemm_max<<<dim3(Bn*S1,2),32,0,stream>>>(gs, Bn*S1, wt[2], kpad[2], F(6), 128,
                                               fb + (size_t)p*Bn*S1*128);
  }

  // 5) sa2 on (pb0,fb0) and (pb1,fb1)
  for (int p=0;p<2;p++){
    const float* pb = l1pos + (size_t)(2*p+1)*Bn*S1*3;
    const float* fbp = fb + (size_t)p*Bn*S1*128;
    int* fi = fps2 + (size_t)p*Bn*S2;
    float* np_ = p2pos + (size_t)p*Bn*S2*3;
    int* gi = gidx2 + (size_t)p*Bn*S2*16;
    k_fps<<<Bn,256,0,stream>>>(pb, S1, S2, fi);
    k_gather_pos<<<cdiv(Bn*S2,256),256,0,stream>>>(pb, fi, np_, S1, S2, Bn*S2);
    k_ballq<<<cdiv(Bn*S2,256),256,0,stream>>>(pb, np_, gi, S1, S2, 1.0f, 16, Bn*S2);
    GSpec gs{};
    gs.pos1=np_; gs.pos2=pb; gs.gidx=gi; gs.Fg=fbp; gs.Fb=nullptr;
    gs.S=S2; gs.n2=S1; gs.ns=16; gs.Cg=128; gs.Cb=0;
    gs.pd_off=0; gs.g_off=3; gs.b_off=0x40000000;
    k_ggemm<<<dim3(Bn*S2,4),32,0,stream>>>(gs, Bn*S2, wt[3], kpad[3], F(8), 256, GA);
    k_gemm<<<dim3(Bn*S2,4),32,0,stream>>>(GA, Bn*S2*16, 256, wt[4], kpad[4], F(10), 256,
                                          f2a + (size_t)p*Bn*S2*256, 0.f, 1);
  }

  // 6) fe2
  {
    const float* P1 = p2pos + (size_t)Bn*S2*3;  // p2a1
    const float* P0 = p2pos;                    // p2a0
    k_knn<<<cdiv(Bn*S2,256),256,0,stream>>>(P0, P1, knn2, S2, S2, 24, Bn*S2);
    GSpec gs{};
    gs.pos1=P1; gs.pos2=P0; gs.gidx=knn2;
    gs.Fg=f2a; gs.Fb=f2a + (size_t)Bn*S2*256;
    gs.S=S2; gs.n2=S2; gs.ns=24; gs.Cg=256; gs.Cb=256;
    gs.pd_off=0; gs.g_off=3; gs.b_off=259;
    k_ggemm_max<<<dim3(Bn*S2,4),32,0,stream>>>(gs, Bn*S2, wt[5], kpad[5], F(12), 256, f2b);
  }

  // 7) sa3 (single layer -> fused max)
  {
    const float* pb = p2pos + (size_t)Bn*S2*3;  // p2b == p2a1
    k_fps<<<Bn,256,0,stream>>>(pb, S2, S3, fps3);
    k_gather_pos<<<cdiv(Bn*S3,256),256,0,stream>>>(pb, fps3, p3pos, S2, S3, Bn*S3);
    k_ballq<<<cdiv(Bn*S3,256),256,0,stream>>>(pb, p3pos, gidx3, S2, S3, 4.0f, 16, Bn*S3);
    GSpec gs{};
    gs.pos1=p3pos; gs.pos2=pb; gs.gidx=gidx3; gs.Fg=f2b; gs.Fb=nullptr;
    gs.S=S3; gs.n2=S2; gs.ns=16; gs.Cg=256; gs.Cb=0;
    gs.pd_off=0; gs.g_off=3; gs.b_off=0x40000000;
    k_ggemm_max<<<dim3(Bn*S3,8),32,0,stream>>>(gs, Bn*S3, wt[6], kpad[6], F(14), 512, f3);
  }

  // 8) su1: upconv 275 -> 1375
  {
    const float* P1 = p2pos + (size_t)Bn*S2*3;
    k_knn<<<cdiv(Bn*S2,256),256,0,stream>>>(p3pos, P1, knnu1, S3, S2, 16, Bn*S2);
    GSpec gs{};
    gs.pos1=P1; gs.pos2=p3pos; gs.gidx=knnu1; gs.Fg=f3; gs.Fb=nullptr;
    gs.S=S2; gs.n2=S3; gs.ns=16; gs.Cg=512; gs.Cb=0;
    gs.pd_off=512; gs.g_off=0; gs.b_off=0x40000000;
    k_ggemm_max<<<dim3(Bn*S2,8),32,0,stream>>>(gs, Bn*S2, wt[7], kpad[7], F(16), 512, hbuf);
    int tot = Bn*S2*768;
    k_concat2<<<cdiv(tot,256),256,0,stream>>>(hbuf, 512, f2a + (size_t)Bn*S2*256, 256, GA, tot, 768);
    k_gemm<<<dim3(cdiv(Bn*S2,16),8),32,0,stream>>>(GA, Bn*S2, 768, wt[8], kpad[8], F(18), 512,
                                                   f2n, 0.f, 0);
  }

  // 9) su2: upconv 1375 -> 5500
  {
    const float* P1 = l1pos + (size_t)3*Bn*S1*3;
    const float* P2 = p2pos + (size_t)Bn*S2*3;
    k_knn<<<cdiv(Bn*S1,256),256,0,stream>>>(P2, P1, knnu2, S2, S1, 16, Bn*S1);
    GSpec gs{};
    gs.pos1=P1; gs.pos2=P2; gs.gidx=knnu2; gs.Fg=f2n; gs.Fb=nullptr;
    gs.S=S1; gs.n2=S2; gs.ns=16; gs.Cg=512; gs.Cb=0;
    gs.pd_off=512; gs.g_off=0; gs.b_off=0x40000000;
    k_ggemm_max<<<dim3(Bn*S1,8),32,0,stream>>>(gs, Bn*S1, wt[9], kpad[9], F(20), 512, hbuf);
    int tot = Bn*S1*640;
    k_concat2<<<cdiv(tot,256),256,0,stream>>>(hbuf, 512, l1ft + (size_t)3*Bn*S1*128, 128, GA, tot, 640);
    k_gemm<<<dim3(cdiv(Bn*S1,16),8),32,0,stream>>>(GA, Bn*S1, 640, wt[10], kpad[10], F(22), 512,
                                                   f1n, 0.f, 0);
  }

  // 10) feature propagation (3-NN interp) + head + final add
  {
    const float* P1 = xs + (size_t)3*Bn*Nn*3;       // xs[3]
    const float* P2 = l1pos + (size_t)3*Bn*S1*3;    // l1 pos of frame 3
    k_fp_interp<<<cdiv(Bn*Nn,256),256,0,stream>>>(P1, P2, f1n, GA, Nn, S1, 512, Bn*Nn);
    k_gemm<<<dim3(Bn*Nn/16,4),32,0,stream>>>(GA, Bn*Nn, 513, wt[11], kpad[11], F(24), 256,
                                             f0b, 0.f, 0);
    k_gemm<<<dim3(Bn*Nn/16,2),32,0,stream>>>(f0b, Bn*Nn, 256, wt[12], kpad[12], F(26), 128,
                                             h1b, 0.2f, 0);
    k_gemm<<<dim3(Bn*Nn/16,1),32,0,stream>>>(h1b, Bn*Nn, 128, wt[13], kpad[13], F(28), 3,
                                             flow, 1.0f, 0);
    k_final<<<cdiv(Bn*3*Nn,256),256,0,stream>>>(x, flow, (float*)d_out, Bn*3*Nn);
  }
}
